// GINConv_51762945852037
// MI455X (gfx1250) — matrix operand order repaired and checked by reading, not compiled, not run on hardware
//
#include <hip/hip_runtime.h>

typedef __attribute__((ext_vector_type(16))) __bf16 v16bf;
typedef __attribute__((ext_vector_type(8)))  float  v8f;
typedef __attribute__((ext_vector_type(4)))  float  v4f;

#define N_NODES 50000
#define E_EDGES 800000
#define DIN 256
#define DOUT 512
#define BN_EPS 1e-5f

// ---------------------------------------------------------------- copy x -> h
__global__ void copy_x_kernel(const float4* __restrict__ x, float4* __restrict__ h) {
    int gid = blockIdx.x * blockDim.x + threadIdx.x;   // exactly N*DIN/4 threads
    h[gid] = x[gid];
}

// ------------------------------------------------- edge scatter: h += w * x[src]
__global__ void scatter_edges_kernel(const float* __restrict__ x,
                                     const int*   __restrict__ ei,
                                     const float* __restrict__ ew,
                                     float* __restrict__ h) {
    long long gid = (long long)blockIdx.x * blockDim.x + threadIdx.x;
    if (gid >= (long long)E_EDGES * (DIN / 4)) return;
    int e  = (int)(gid >> 6);            // DIN/4 == 64 chunks per edge
    int fo = ((int)gid & 63) * 4;
    int dst = ei[e];                      // edge_index[0][e]
    int src = ei[E_EDGES + e];            // edge_index[1][e]
    float w = ew[e];
    const float4 xv = *(const float4*)(x + (size_t)src * DIN + fo);
    float* hp = h + (size_t)dst * DIN + fo;
    atomicAdd(hp + 0, w * xv.x);
    atomicAdd(hp + 1, w * xv.y);
    atomicAdd(hp + 2, w * xv.z);
    atomicAdd(hp + 3, w * xv.w);
}

// ------------------------------------------- pack W (f32 KxN) -> bf16 hi/lo B-fragments
// Fragment f = kt*ntiles + nt ; lane l holds B[kt*32 + 16*(l>>4) + j][nt*16 + (l&15)]
// stored contiguously: packed[(f*32 + l)*16 + j]
__global__ void pack_w_kernel(const float* __restrict__ W,
                              __bf16* __restrict__ hi, __bf16* __restrict__ lo,
                              int K, int Nc) {
    int t = blockIdx.x * blockDim.x + threadIdx.x;
    int ntiles = Nc >> 4;
    int total  = (K >> 5) * ntiles * 32;
    if (t >= total) return;
    int lane = t & 31;
    int f    = t >> 5;
    int kt   = f / ntiles, nt = f % ntiles;
    int krow0 = kt * 32 + (lane >> 4) * 16;
    int col   = nt * 16 + (lane & 15);
#pragma unroll
    for (int j = 0; j < 16; ++j) {
        float v = W[(size_t)(krow0 + j) * Nc + col];
        __bf16 hb = (__bf16)v;
        hi[(size_t)t * 16 + j] = hb;
        lo[(size_t)t * 16 + j] = (__bf16)(v - (float)hb);
    }
}

// ------------------------------------------------------- zero helper (stats)
__global__ void zero_f_kernel(float* __restrict__ p) {
    p[blockIdx.x * blockDim.x + threadIdx.x] = 0.0f;
}

// ---------------------------------------------- C = A(f32, MxK) * B(packed) + bias
// bf16x3 split-precision WMMA GEMM. Block = 256 thr = 8 waves; wave w owns 16x64 slab.
__global__ void __launch_bounds__(256)
gemm_bf16x3_kernel(const float* __restrict__ A,
                   const __bf16* __restrict__ Bhi, const __bf16* __restrict__ Blo,
                   const float* __restrict__ bias, float* __restrict__ C,
                   int K, int Nc) {
    const int lane = threadIdx.x & 31;
    const int wave = threadIdx.x >> 5;
    const int row0 = blockIdx.x * 16;
    const int lrow = lane & 15;   // M offset in A-frag / N offset in B,C frags
    const int lhi  = lane >> 4;   // lane group (0,1)
    const int ntiles = Nc >> 4;
    const int ktiles = K >> 5;
    const int nt0 = wave * 4;

    v8f acc[4];
#pragma unroll
    for (int t = 0; t < 4; ++t) {
        float bv = bias[(nt0 + t) * 16 + lrow];
#pragma unroll
        for (int r = 0; r < 8; ++r) acc[t][r] = bv;
    }

    const float* arow = A + (size_t)(row0 + lrow) * K;
    const v16bf* bhiP = (const v16bf*)Bhi;
    const v16bf* bloP = (const v16bf*)Blo;

    for (int kt = 0; kt < ktiles; ++kt) {
        // A fragment (contiguous k order, matches packed B): elems 0..15 = K[kbase..kbase+15],
        // kbase = kt*32 + 16*(lane>>4)
        const int kbase = kt * 32 + lhi * 16;
        v4f a0 = *(const v4f*)(arow + kbase);
        v4f a1 = *(const v4f*)(arow + kbase + 4);
        v4f a2 = *(const v4f*)(arow + kbase + 8);
        v4f a3 = *(const v4f*)(arow + kbase + 12);
        v16bf ahi, alo;
#pragma unroll
        for (int j = 0; j < 4; ++j) {
            float f;
            __bf16 hb;
            f = a0[j]; hb = (__bf16)f; ahi[j]      = hb; alo[j]      = (__bf16)(f - (float)hb);
            f = a1[j]; hb = (__bf16)f; ahi[j + 4]  = hb; alo[j + 4]  = (__bf16)(f - (float)hb);
            f = a2[j]; hb = (__bf16)f; ahi[j + 8]  = hb; alo[j + 8]  = (__bf16)(f - (float)hb);
            f = a3[j]; hb = (__bf16)f; ahi[j + 12] = hb; alo[j + 12] = (__bf16)(f - (float)hb);
        }
#pragma unroll
        for (int t = 0; t < 4; ++t) {
            int f = kt * ntiles + (nt0 + t);
            v16bf bh = bhiP[(size_t)f * 32 + lane];
            v16bf bl = bloP[(size_t)f * 32 + lane];
            // small terms first, fp32 accumulate; a_lo*b_lo term dropped (~2^-18)
            acc[t] = __builtin_amdgcn_wmma_f32_16x16x32_bf16(false, alo, false, bh, (short)0, acc[t], false, false);
            acc[t] = __builtin_amdgcn_wmma_f32_16x16x32_bf16(false, ahi, false, bl, (short)0, acc[t], false, false);
            acc[t] = __builtin_amdgcn_wmma_f32_16x16x32_bf16(false, ahi, false, bh, (short)0, acc[t], false, false);
        }
    }

#pragma unroll
    for (int t = 0; t < 4; ++t) {
        int col = (nt0 + t) * 16 + lrow;
#pragma unroll
        for (int r = 0; r < 8; ++r) {
            int row = row0 + r + lhi * 8;   // C/D layout: VGPR r = M=r (lanes0-15), M=8+r (16-31)
            C[(size_t)row * Nc + col] = acc[t][r];
        }
    }
}

// ------------------------------------------------- per-column sums / sumsq
__global__ void __launch_bounds__(512)
bn_stats_kernel(const float* __restrict__ h1, float* __restrict__ sums,
                float* __restrict__ sumsq) {
    int col = threadIdx.x;             // 0..511
    int r0  = blockIdx.x * 128;
    float s = 0.0f, q = 0.0f;
    for (int i = 0; i < 128; ++i) {
        int r = r0 + i;
        if (r < N_NODES) {
            float v = h1[(size_t)r * DOUT + col];
            s += v; q += v * v;
        }
    }
    atomicAdd(&sums[col], s);
    atomicAdd(&sumsq[col], q);
}

// ------------------------------------------------- in-place BN (train stats) + ReLU
__global__ void bn_relu_kernel(float* __restrict__ h1, const float* __restrict__ sums,
                               const float* __restrict__ sumsq,
                               const float* __restrict__ gamma,
                               const float* __restrict__ beta) {
    long long gid = (long long)blockIdx.x * blockDim.x + threadIdx.x;
    if (gid >= (long long)N_NODES * DOUT) return;
    int col = (int)(gid & (DOUT - 1));
    const float invN = 1.0f / (float)N_NODES;
    float mean  = sums[col] * invN;
    float var   = sumsq[col] * invN - mean * mean;   // biased, matches jnp.var
    float scale = gamma[col] * rsqrtf(var + BN_EPS);
    float v = (h1[gid] - mean) * scale + beta[col];
    h1[gid] = v > 0.0f ? v : 0.0f;
}

extern "C" void kernel_launch(void* const* d_in, const int* in_sizes, int n_in,
                              void* d_out, int out_size, void* d_ws, size_t ws_size,
                              hipStream_t stream) {
    const float* x     = (const float*)d_in[0];
    const int*   ei    = (const int*)  d_in[1];
    const float* ew    = (const float*)d_in[2];
    const float* W1    = (const float*)d_in[3];
    const float* b1    = (const float*)d_in[4];
    const float* gamma = (const float*)d_in[5];
    const float* beta  = (const float*)d_in[6];
    const float* W2    = (const float*)d_in[7];
    const float* b2    = (const float*)d_in[8];
    float* out = (float*)d_out;

    char* ws = (char*)d_ws;                       // all offsets 256B-aligned
    float*  h     = (float*)(ws + 0);             // N*DIN  f32 = 51,200,000 B
    float*  h1    = (float*)(ws + 51200000);      // N*DOUT f32 = 102,400,000 B
    float*  sums  = (float*)(ws + 153600000);     // 512 f32
    float*  sumsq = (float*)(ws + 153602048);     // 512 f32
    __bf16* w1hi  = (__bf16*)(ws + 153604096);    // 256*512 bf16 = 262,144 B
    __bf16* w1lo  = (__bf16*)(ws + 153866240);
    __bf16* w2hi  = (__bf16*)(ws + 154128384);    // 512*512 bf16 = 524,288 B
    __bf16* w2lo  = (__bf16*)(ws + 154652672);    // end @ 155,176,960 B

    // h = (1+eps)*x = x   (eps==0)
    copy_x_kernel<<<(N_NODES * DIN / 4) / 256, 256, 0, stream>>>(
        (const float4*)x, (float4*)h);

    // weight packing + stats zeroing (independent of h)
    pack_w_kernel<<<(8 * 32 * 32 + 255) / 256, 256, 0, stream>>>(W1, w1hi, w1lo, DIN, DOUT);
    pack_w_kernel<<<(16 * 32 * 32 + 255) / 256, 256, 0, stream>>>(W2, w2hi, w2lo, DOUT, DOUT);
    zero_f_kernel<<<4, 256, 0, stream>>>(sums);   // zeros sums(512) + sumsq(512)

    // h += scatter_add(edge_weight * x[col])
    {
        long long work = (long long)E_EDGES * (DIN / 4);
        int blocks = (int)((work + 255) / 256);
        scatter_edges_kernel<<<blocks, 256, 0, stream>>>(x, ei, ew, h);
    }

    // h1 = h @ W1 + b1      (M=50000, K=256, N=512)
    gemm_bf16x3_kernel<<<N_NODES / 16, 256, 0, stream>>>(h, w1hi, w1lo, b1, h1, DIN, DOUT);

    // BN stats + in-place BN/ReLU
    bn_stats_kernel<<<(N_NODES + 127) / 128, 512, 0, stream>>>(h1, sums, sumsq);
    {
        long long work = (long long)N_NODES * DOUT;
        int blocks = (int)((work + 255) / 256);
        bn_relu_kernel<<<blocks, 256, 0, stream>>>(h1, sums, sumsq, gamma, beta);
    }

    // out = h1 @ W2 + b2    (M=50000, K=512, N=512)
    gemm_bf16x3_kernel<<<N_NODES / 16, 256, 0, stream>>>(h1, w2hi, w2lo, b2, out, DOUT, DOUT);
}
